// CasualAttention_5128190951601
// MI455X (gfx1250) — matrix operand order repaired and checked by reading, not compiled, not run on hardware
//
#include <hip/hip_runtime.h>

// ---------------------------------------------------------------------------
// CDNA5 (gfx1250) causal attention: fp32 in/out, bf16 WMMA compute.
//   B=4, S=2048, D=1024.  Q/K stored bf16 [B*S, D]; V stored bf16 [B, D, S].
// ---------------------------------------------------------------------------

#define BATCH 4
#define SEQ   2048
#define DIM   1024
#define MROWS (BATCH * SEQ)   // 8192

typedef __bf16 bf16_t;
typedef bf16_t bf16x8  __attribute__((ext_vector_type(8)));
typedef bf16_t bf16x16 __attribute__((ext_vector_type(16)));
typedef float  f32x4   __attribute__((ext_vector_type(4)));
typedef float  f32x8   __attribute__((ext_vector_type(8)));
typedef int    i32x4   __attribute__((ext_vector_type(4)));

#define AS1 __attribute__((address_space(1)))
#define AS3 __attribute__((address_space(3)))

#if __has_builtin(__builtin_amdgcn_global_load_async_to_lds_b128)
#define HAVE_ASYNC_LDS 1
#endif

static __device__ __forceinline__ void wait_asynccnt0() {
#if __has_builtin(__builtin_amdgcn_s_wait_asynccnt)
    __builtin_amdgcn_s_wait_asynccnt(0);
#else
    asm volatile("s_wait_asynccnt 0x0" ::: "memory");
#endif
}

static __device__ __forceinline__ f32x8 wmma_bf16(bf16x16 a, bf16x16 b, f32x8 c) {
    return __builtin_amdgcn_wmma_f32_16x16x32_bf16(false, a, false, b, (short)0, c,
                                                   false, false);
}

// Assemble a v16bf A/B fragment from two 16-byte chunks (16B-aligned LDS safe).
static __device__ __forceinline__ bf16x16 frag2(const bf16_t* p0, const bf16_t* p1) {
    bf16x8 lo = *(const bf16x8*)p0;
    bf16x8 hi = *(const bf16x8*)p1;
    return __builtin_shufflevector(lo, hi, 0,1,2,3,4,5,6,7,8,9,10,11,12,13,14,15);
}

// ---------------------------------------------------------------------------
// Kernel 1: fused QKV projection.  y[m,n] = sum_k x[m,k] * W[n,k]  (x @ W^T)
// grid = (N/128, M/128, 3); block = 256 (8 wave32).
// z==0 -> Q [row][col], z==1 -> K [row][col], z==2 -> V transposed [b][col][s]
// ---------------------------------------------------------------------------
__global__ __launch_bounds__(256) void qkv_proj_wmma(
    const float* __restrict__ X,
    const float* __restrict__ Wq, const float* __restrict__ Wk,
    const float* __restrict__ Wv,
    bf16_t* __restrict__ Qo, bf16_t* __restrict__ Ko, bf16_t* __restrict__ Vto)
{
    __shared__ bf16_t Xs[128][40];   // 128 rows x 32 k (+8 pad), 80B row stride
    __shared__ bf16_t Ws[128][40];

    const int tid  = threadIdx.x;
    const int lane = tid & 31;
    const int wave = tid >> 5;
    const int l15  = lane & 15;
    const int hi   = lane >> 4;          // which 16-lane half

    const int z  = blockIdx.z;
    const float* W = (z == 0) ? Wq : ((z == 1) ? Wk : Wv);
    const int n0 = blockIdx.x * 128;
    const int m0 = blockIdx.y * 128;

    const int wm = wave & 1;             // 64-row half of the 128x128 tile
    const int wn = wave >> 1;            // 32-col group

    f32x8 acc[4][2] = {};

    for (int kt = 0; kt < DIM / 32; ++kt) {
        __syncthreads();
        // ---- stage 128x32 fp32->bf16 tiles of X and W into LDS -------------
        #pragma unroll
        for (int i = 0; i < 4; ++i) {
            int idx = tid + i * 256;          // 0..1023 float4 slots
            int row = idx >> 3;
            int kq  = (idx & 7) * 4;
            f32x4 xv = *(const f32x4*)(X + (size_t)(m0 + row) * DIM + kt * 32 + kq);
            bf16_t* dx = &Xs[row][kq];
            dx[0] = (bf16_t)xv[0]; dx[1] = (bf16_t)xv[1];
            dx[2] = (bf16_t)xv[2]; dx[3] = (bf16_t)xv[3];
            f32x4 wv = *(const f32x4*)(W + (size_t)(n0 + row) * DIM + kt * 32 + kq);
            bf16_t* dw = &Ws[row][kq];
            dw[0] = (bf16_t)wv[0]; dw[1] = (bf16_t)wv[1];
            dw[2] = (bf16_t)wv[2]; dw[3] = (bf16_t)wv[3];
        }
        // prefetch next K-chunk into near caches (global_prefetch_b8)
        if (kt + 1 < DIM / 32) {
            __builtin_prefetch(X + (size_t)(m0 + (tid >> 1)) * DIM + (kt + 1) * 32, 0, 3);
            __builtin_prefetch(W + (size_t)(n0 + (tid >> 1)) * DIM + (kt + 1) * 32, 0, 3);
        }
        __syncthreads();

        // ---- B fragments: lane holds col N = wn*32+nf*16+l15, K = hi*16.. --
        bf16x16 bfr[2];
        #pragma unroll
        for (int nf = 0; nf < 2; ++nf) {
            const bf16_t* bp = &Ws[wn * 32 + nf * 16 + l15][hi * 16];
            bfr[nf] = frag2(bp, bp + 8);
        }
        // ---- A fragments + 8 WMMAs --------------------------------------
        #pragma unroll
        for (int mf = 0; mf < 4; ++mf) {
            const bf16_t* ap = &Xs[wm * 64 + mf * 16 + l15][0];
            bf16x16 afr = frag2(ap + hi * 16, ap + hi * 16 + 8);
            acc[mf][0] = wmma_bf16(afr, bfr[0], acc[mf][0]);
            acc[mf][1] = wmma_bf16(afr, bfr[1], acc[mf][1]);
        }
    }

    // ---- epilogue: fp32 acc -> bf16 stores ------------------------------
    #pragma unroll
    for (int mf = 0; mf < 4; ++mf)
        #pragma unroll
        for (int nf = 0; nf < 2; ++nf)
            #pragma unroll
            for (int r = 0; r < 8; ++r) {
                int row = m0 + wm * 64 + mf * 16 + hi * 8 + r;
                int col = n0 + wn * 32 + nf * 16 + l15;
                bf16_t v = (bf16_t)acc[mf][nf][r];
                if (z == 0) {
                    Qo[(size_t)row * DIM + col] = v;
                } else if (z == 1) {
                    Ko[(size_t)row * DIM + col] = v;
                } else {
                    int bb = row >> 11;          // / SEQ
                    int ss = row & (SEQ - 1);
                    Vto[((size_t)bb * DIM + col) * SEQ + ss] = v;
                }
            }
}

// ---------------------------------------------------------------------------
// Kernel 2: flash attention with causal mask.
// One workgroup per (b, 32-query tile). 256 threads = 8 wave32.
// Per 128-key block: stage1 scores (8x 16x32 wave jobs), softmax, stage2 P@V
// with d split 8x128 across waves.  Dynamic LDS = 92,032 B.
// ---------------------------------------------------------------------------
#define QS_STRIDE 1032           // halfs per Q row in LDS (bank-conflict pad)
#define S_STRIDE  132            // floats per score row
#define P_STRIDE  136            // halfs per P row
#define QS_BYTES  (32 * QS_STRIDE * 2)
#define SL_BYTES  (32 * S_STRIDE * 4)
#define PL_BYTES  (32 * P_STRIDE * 2)
#define ATTN_LDS  (QS_BYTES + SL_BYTES + PL_BYTES + 3 * 32 * 4)

__global__ __launch_bounds__(256) void attn_flash_wmma(
    const bf16_t* __restrict__ Qg, const bf16_t* __restrict__ Kg,
    const bf16_t* __restrict__ Vtg, float* __restrict__ Out)
{
    constexpr float SCALE = 0.03125f;              // 1/sqrt(1024)
    constexpr float L2E   = 1.4426950408889634f;   // log2(e)

    extern __shared__ char lds_raw[];
    bf16_t* Qs    = (bf16_t*)lds_raw;                               // [32][1032]
    float*  Sld   = (float*)(lds_raw + QS_BYTES);                   // [32][132]
    bf16_t* Pld   = (bf16_t*)(lds_raw + QS_BYTES + SL_BYTES);       // [32][136]
    float*  m_run = (float*)(lds_raw + QS_BYTES + SL_BYTES + PL_BYTES);
    float*  l_run = m_run + 32;
    float*  a_lds = m_run + 64;

    const int tid  = threadIdx.x;
    const int lane = tid & 31;
    const int wave = tid >> 5;
    const int l15  = lane & 15;
    const int hi   = lane >> 4;

    const int qtiles = SEQ / 32;
    const int b  = blockIdx.x / qtiles;
    const int q0 = (blockIdx.x % qtiles) * 32;

    // ---- stage the 32x1024 bf16 Q tile into LDS --------------------------
    // Preferred path: gfx1250 async global->LDS DMA (tracked by ASYNCcnt),
    // which keeps the copy out of the VGPR file. Fallback: uint4 copies.
    const bf16_t* Qbase = Qg + ((size_t)b * SEQ + q0) * DIM;
#ifdef HAVE_ASYNC_LDS
    for (int i = tid; i < 32 * 128; i += 256) {       // 128 x 16B chunks / row
        int row = i >> 7;
        int c16 = i & 127;
        const char* src = (const char*)(Qbase + (size_t)row * DIM) + c16 * 16;
        char*       dst = (char*)(Qs + row * QS_STRIDE) + c16 * 16;
        __builtin_amdgcn_global_load_async_to_lds_b128(
            (AS1 i32x4*)(uintptr_t)src,
            (AS3 i32x4*)(uint32_t)(uintptr_t)dst, 0, 0);
    }
    wait_asynccnt0();
#else
    for (int i = tid; i < 32 * 128; i += 256) {       // 128 x 16B chunks / row
        int row = i >> 7;
        int c16 = i & 127;
        *(uint4*)((char*)(Qs + row * QS_STRIDE) + c16 * 16) =
            *(const uint4*)((const char*)(Qbase + (size_t)row * DIM) + c16 * 16);
    }
#endif
    if (tid < 32) { m_run[tid] = -1e30f; l_run[tid] = 0.0f; }
    __syncthreads();

    f32x8 o[2][8] = {};      // per-wave output slice: 32 rows x 128 cols fp32

    const int mf1 = wave & 1;       // stage-1: 16-row group
    const int nj  = wave >> 1;      // stage-1: 32-key group within 128
    const int nkb = q0 / 128 + 1;   // causal key-block count

    for (int kb = 0; kb < nkb; ++kb) {
        const int k0 = kb * 128;

        // ================= stage 1: S = Q * K^T  (16x32 per wave) ==========
        f32x8 s0 = {}, s1 = {};
        const bf16_t* Kb0 = Kg + ((size_t)b * SEQ + k0 + nj * 32 + l15) * DIM;
        const bf16_t* Kb1 = Kb0 + (size_t)16 * DIM;
        const bf16_t* arow = Qs + (mf1 * 16 + l15) * QS_STRIDE;
        #pragma unroll 4
        for (int dc = 0; dc < DIM / 32; ++dc) {
            bf16x16 af = frag2(arow + dc * 32 + hi * 16, arow + dc * 32 + hi * 16 + 8);
            bf16x16 b0 = *(const bf16x16*)(Kb0 + dc * 32 + hi * 16);
            bf16x16 b1 = *(const bf16x16*)(Kb1 + dc * 32 + hi * 16);
            s0 = wmma_bf16(af, b0, s0);
            s1 = wmma_bf16(af, b1, s1);
        }
        {
            int rowb = mf1 * 16 + hi * 8;
            int colb = nj * 32 + l15;
            #pragma unroll
            for (int r = 0; r < 8; ++r) {
                Sld[(rowb + r) * S_STRIDE + colb]      = s0[r];
                Sld[(rowb + r) * S_STRIDE + colb + 16] = s1[r];
            }
        }
        __syncthreads();

        // ================= softmax (online, causal) ========================
        {
            int row = wave * 4 + (lane >> 3);   // 8 lanes per row
            int sub = lane & 7;
            int c0  = sub * 16;
            int qg  = q0 + row;
            float sv[16];
            float mx = -1e30f;
            #pragma unroll
            for (int i = 0; i < 16; ++i) {
                int kgl = k0 + c0 + i;
                float v = (kgl <= qg) ? Sld[row * S_STRIDE + c0 + i] * SCALE : -1e30f;
                sv[i] = v;
                mx = fmaxf(mx, v);
            }
            mx = fmaxf(mx, __shfl_xor(mx, 1));
            mx = fmaxf(mx, __shfl_xor(mx, 2));
            mx = fmaxf(mx, __shfl_xor(mx, 4));
            float mo = m_run[row];
            float mn = fmaxf(mo, mx);
            float alpha = exp2f((mo - mn) * L2E);
            float sum = 0.0f;
            #pragma unroll
            for (int i = 0; i < 16; ++i) {
                float e = (sv[i] > -1e29f) ? exp2f((sv[i] - mn) * L2E) : 0.0f;
                sum += e;
                Pld[row * P_STRIDE + c0 + i] = (bf16_t)e;
            }
            sum += __shfl_xor(sum, 1);
            sum += __shfl_xor(sum, 2);
            sum += __shfl_xor(sum, 4);
            if (sub == 0) {
                m_run[row] = mn;
                l_run[row] = l_run[row] * alpha + sum;
                a_lds[row] = alpha;
            }
        }
        __syncthreads();

        // ================= stage 2: O = O*alpha + P @ V_slice ==============
        #pragma unroll
        for (int mf = 0; mf < 2; ++mf) {
            f32x8 al = *(const f32x8*)(a_lds + mf * 16 + hi * 8);
            #pragma unroll
            for (int nf = 0; nf < 8; ++nf)
                #pragma unroll
                for (int r = 0; r < 8; ++r)
                    o[mf][nf][r] *= al[r];
        }
        const bf16_t* Vb = Vtg + ((size_t)b * DIM + wave * 128 + l15) * SEQ + k0;
        #pragma unroll
        for (int kc = 0; kc < 4; ++kc) {
            const bf16_t* pr0 = Pld + (size_t)l15 * P_STRIDE + kc * 32;
            const bf16_t* pr1 = Pld + (size_t)(16 + l15) * P_STRIDE + kc * 32;
            bf16x16 p0 = frag2(pr0 + hi * 16, pr0 + hi * 16 + 8);
            bf16x16 p1 = frag2(pr1 + hi * 16, pr1 + hi * 16 + 8);
            #pragma unroll
            for (int nf = 0; nf < 8; ++nf) {
                bf16x16 bv = *(const bf16x16*)(Vb + (size_t)nf * 16 * SEQ + kc * 32 + hi * 16);
                o[0][nf] = wmma_bf16(p0, bv, o[0][nf]);
                o[1][nf] = wmma_bf16(p1, bv, o[1][nf]);
            }
        }
        __syncthreads();   // protect S/P/alpha before next block overwrites
    }

    // ---- epilogue: normalize by row sums, store fp32 ---------------------
    #pragma unroll
    for (int mf = 0; mf < 2; ++mf) {
        f32x8 lv = *(const f32x8*)(l_run + mf * 16 + hi * 8);
        f32x8 inv;
        #pragma unroll
        for (int r = 0; r < 8; ++r) inv[r] = 1.0f / lv[r];
        #pragma unroll
        for (int nf = 0; nf < 8; ++nf)
            #pragma unroll
            for (int r = 0; r < 8; ++r) {
                int row = q0 + mf * 16 + hi * 8 + r;
                int col = wave * 128 + nf * 16 + l15;
                Out[((size_t)b * SEQ + row) * DIM + col] = o[mf][nf][r] * inv[r];
            }
    }
}

// ---------------------------------------------------------------------------
extern "C" void kernel_launch(void* const* d_in, const int* in_sizes, int n_in,
                              void* d_out, int out_size, void* d_ws, size_t ws_size,
                              hipStream_t stream) {
    const float* X  = (const float*)d_in[0];
    const float* Wq = (const float*)d_in[1];
    const float* Wk = (const float*)d_in[2];
    const float* Wv = (const float*)d_in[3];
    float* Out = (float*)d_out;

    // workspace: Q bf16 [8192,1024] | K bf16 [8192,1024] | Vt bf16 [4,1024,2048]
    bf16_t* Qw  = (bf16_t*)d_ws;
    bf16_t* Kw  = Qw + (size_t)MROWS * DIM;
    bf16_t* Vtw = Kw + (size_t)MROWS * DIM;

    dim3 g1(DIM / 128, MROWS / 128, 3);
    qkv_proj_wmma<<<g1, 256, 0, stream>>>(X, Wq, Wk, Wv, Qw, Kw, Vtw);

    static_assert(ATTN_LDS == 92032, "lds layout");
    (void)hipFuncSetAttribute((const void*)attn_flash_wmma,
                              hipFuncAttributeMaxDynamicSharedMemorySize, ATTN_LDS);
    dim3 g2(BATCH * (SEQ / 32));
    attn_flash_wmma<<<g2, 256, ATTN_LDS, stream>>>(Qw, Kw, Vtw, Out);
}
